// StructuredBlock_74483322847386
// MI455X (gfx1250) — compile-verified, matrix operand order checked by reading, not run on hardware
//
#include <hip/hip_runtime.h>
#include <math.h>

// ---------------- constants ----------------
#define HH   48
#define WW   48
#define WSZ  8
#define NHD  12
#define CCH  768
#define DH   64
#define LTK  (HH*WW)          // 2304
#define BB   2
#define NTOK (BB*LTK)         // 4608
#define MFF  3072

// ---------------- WMMA types ----------------
typedef __attribute__((ext_vector_type(16))) __bf16 v16bf;
typedef __attribute__((ext_vector_type(8)))  float  v8f;

union AF { v16bf v; unsigned u[8]; };

static __device__ inline v8f wmma_bf16(const AF& a, const AF& b, v8f c) {
  return __builtin_amdgcn_wmma_f32_16x16x32_bf16(false, a.v, false, b.v,
                                                 (short)0, c, false, false);
}

// K-pair start for VGPR v, lane-half h (16-bit A/B fragment layout, 16x16x32)
static __device__ inline int kpair(int v, int h) {
  return ((v < 4) ? 0 : 16) + 8 * h + 2 * (v & 3);
}

static __device__ inline unsigned short f2bf(float f) {
  unsigned u = __float_as_uint(f);
  unsigned r = u + 0x7FFFu + ((u >> 16) & 1u);
  return (unsigned short)(r >> 16);
}

// ---------------- gfx1250 async global->LDS copy (16B), guarded ----------------
#define AS1 __attribute__((address_space(1)))
#define AS3 __attribute__((address_space(3)))

typedef int v4i_ __attribute__((ext_vector_type(4)));

#if __has_builtin(__builtin_amdgcn_global_load_async_to_lds_b128)
#define HAVE_ASYNC_LDS 1
#endif

static __device__ inline void cp16_to_lds(void* l, const void* g) {
#ifdef HAVE_ASYNC_LDS
  // param types per clang: (v4i AS1*, v4i AS3*, int imm_offset, int cpol)
  __builtin_amdgcn_global_load_async_to_lds_b128(
      (AS1 v4i_*)g, (AS3 v4i_*)l, 0, 0);
#else
  *(uint4*)l = *(const uint4*)g;
#endif
}

static __device__ inline void async_fence() {
#ifdef HAVE_ASYNC_LDS
#if __has_builtin(__builtin_amdgcn_s_wait_asynccnt)
  __builtin_amdgcn_s_wait_asynccnt(0);
#else
  asm volatile("s_wait_asynccnt 0" ::: "memory");
#endif
#endif
}

// ---------------- fp32 -> bf16 convert ----------------
__global__ void __launch_bounds__(256) cvt_bf16_kernel(const float* __restrict__ in,
                                                       unsigned short* __restrict__ out,
                                                       int n) {
  int i = blockIdx.x * 256 + threadIdx.x;
  if (i < n) out[i] = f2bf(in[i]);
}

// ---------------- mask / selector ----------------
// one 64-thread block per 8x8 window; sel: 2=fg(global), 1=bg-window(local), 0=zero
__global__ void __launch_bounds__(64) sel_kernel(const float* __restrict__ mask,
                                                 float* __restrict__ sel) {
  int blk = blockIdx.x;
  int b = blk / 36, win = blk % 36;
  int wy = win / 6, wx = win % 6;
  int t = threadIdx.x;
  int h = wy * WSZ + (t >> 3);
  int w = wx * WSZ + (t & 7);
  const float* mb = mask + (size_t)b * (4 * HH) * (4 * WW);
  float s = 0.f;
#pragma unroll
  for (int r = 0; r < 4; ++r)
#pragma unroll
    for (int c = 0; c < 4; ++c)
      s += mb[(size_t)(4 * h + r) * (4 * WW) + 4 * w + c];
  int fg = (s * (1.f / 16.f)) > 0.4f;

  __shared__ int anybg;
  if (t == 0) anybg = 0;
  __syncthreads();
  if (!fg) anybg = 1;
  __syncthreads();
  sel[(size_t)b * LTK + h * WW + w] = fg ? 2.f : (anybg ? 1.f : 0.f);
}

// ---------------- LayerNorm (row per block), bf16 out ----------------
__global__ void __launch_bounds__(256) ln_kernel(const float* __restrict__ x,
                                                 const float* __restrict__ w,
                                                 const float* __restrict__ bia,
                                                 unsigned short* __restrict__ out16) {
  __shared__ float s1[256], s2[256];
  int row = blockIdx.x, tid = threadIdx.x;
  const float* xr = x + (size_t)row * CCH;
  float v0 = xr[tid], v1 = xr[tid + 256], v2 = xr[tid + 512];
  s1[tid] = v0 + v1 + v2;
  s2[tid] = v0 * v0 + v1 * v1 + v2 * v2;
  __syncthreads();
  for (int o = 128; o > 0; o >>= 1) {
    if (tid < o) { s1[tid] += s1[tid + o]; s2[tid] += s2[tid + o]; }
    __syncthreads();
  }
  float mu = s1[0] * (1.f / CCH);
  float var = s2[0] * (1.f / CCH) - mu * mu;
  float rs = rsqrtf(var + 1e-6f);
  unsigned short* orow = out16 + (size_t)row * CCH;
  orow[tid]       = f2bf((v0 - mu) * rs * w[tid]       + bia[tid]);
  orow[tid + 256] = f2bf((v1 - mu) * rs * w[tid + 256] + bia[tid + 256]);
  orow[tid + 512] = f2bf((v2 - mu) * rs * w[tid + 512] + bia[tid + 512]);
}

// ---------------- tiled WMMA GEMM: C = A(MxK) * B(KxN) (+bias,+gelu,+resid) ----------------
#define BM 64
#define BN 64
#define BK 32
__global__ void __launch_bounds__(256) gemm_bf16_wmma(
    const unsigned short* __restrict__ A, const unsigned short* __restrict__ Bm,
    const float* __restrict__ bias, const float* __restrict__ resid,
    float* __restrict__ Cf, unsigned short* __restrict__ Cb,
    int Mrows, int N, int K, int doGelu) {
  (void)Mrows;
  __shared__ unsigned short As[BM][BK + 8];   // 80B row stride: 16B aligned groups
  __shared__ unsigned short Bt[BN][BK + 8];   // B tile stored transposed [n][k]
  int tid = threadIdx.x;
  int lane = tid & 31, wid = tid >> 5;
  int wm = wid & 3, wn = wid >> 2;
  int half = lane >> 4, l15 = lane & 15;
  int bm0 = blockIdx.y * BM, bn0 = blockIdx.x * BN;

  int arow = tid >> 2, acol = (tid & 3) * 8;  // 64 x 32 tile, 16B per thread
  int brow = tid >> 3, bcol = (tid & 7) * 8;  // 32 x 64 tile, 16B per thread

  v8f acc0 = {}, acc1 = {};
  for (int kt = 0; kt < K; kt += BK) {
    // A tile: async 16B global->LDS (row-major, matches A-fragment layout)
    cp16_to_lds(&As[arow][acol], &A[(size_t)(bm0 + arow) * K + kt + acol]);
    // B tile: load 8 n-contiguous values, scatter transposed into Bt[n][k]
    uint4 bv = *(const uint4*)&Bm[(size_t)(kt + brow) * N + bn0 + bcol];
    const unsigned short* bs = (const unsigned short*)&bv;
#pragma unroll
    for (int j = 0; j < 8; ++j) Bt[bcol + j][brow] = bs[j];
    if (kt + BK < K)
      __builtin_prefetch(&Bm[(size_t)(kt + BK + brow) * N + bn0 + bcol], 0, 1);
    async_fence();
    __syncthreads();

    AF af, bf0, bf1;
    int am = wm * 16 + l15;
    int bn = wn * 32 + l15;
#pragma unroll
    for (int v = 0; v < 8; ++v) {
      int k0 = kpair(v, half);
      af.u[v]  = *(const unsigned*)&As[am][k0];
      bf0.u[v] = *(const unsigned*)&Bt[bn][k0];
      bf1.u[v] = *(const unsigned*)&Bt[bn + 16][k0];
    }
    acc0 = wmma_bf16(af, bf0, acc0);
    acc1 = wmma_bf16(af, bf1, acc1);
    __syncthreads();
  }

#pragma unroll
  for (int r = 0; r < 8; ++r) {
    int m = bm0 + wm * 16 + r + 8 * half;
#pragma unroll
    for (int nt = 0; nt < 2; ++nt) {
      int n = bn0 + wn * 32 + nt * 16 + l15;
      float v = nt ? acc1[r] : acc0[r];
      if (bias) v += bias[n];
      if (doGelu) v = 0.5f * v * (1.f + erff(v * 0.70710678118f));
      size_t idx = (size_t)m * N + n;
      if (resid) v += resid[idx];
      if (Cf) Cf[idx] = v;
      if (Cb) Cb[idx] = f2bf(v);
    }
  }
}

// ---------------- flash-style WMMA attention (global or windowed) ----------------
// qkv: [3][NTOK][C] bf16.  Each wave owns a 16-query tile; keys in 32-chunks.
// Global: all 8 waves share (b,h) -> K/V chunks staged once per block in LDS.
template <bool LOCAL>
__global__ void __launch_bounds__(256) attn_kernel(
    const unsigned short* __restrict__ qkv, const float* __restrict__ sel,
    unsigned short* __restrict__ ctx) {
  __shared__ unsigned short pbuf[8][16][36];  // wave-private P staging
  __shared__ unsigned short Ks[32][72];       // K chunk [key][d]   (144B stride)
  __shared__ unsigned short Vt[64][40];       // V chunk [d][key]   (80B stride)
  int tid = threadIdx.x, lane = tid & 31, wid = tid >> 5;
  int half = lane >> 4, l15 = lane & 15;

  int b, h, q0, win = 0, nkeys;
  if (LOCAL) {
    int task = blockIdx.x * 8 + wid;            // 72 win * 12 heads * 4 qtiles = 3456
    q0 = (task & 3) * 16;
    h = (task >> 2) % NHD;
    win = (task / 48) % 36;
    b = task / (48 * 36);
    nkeys = WSZ * WSZ;
  } else {
    int qt = blockIdx.x * 8 + wid;              // 144 qtiles per (b,h)
    q0 = qt * 16;
    h = blockIdx.y; b = blockIdx.z;
    nkeys = LTK;
  }
  int wy = win / 6, wx = win % 6;
  auto tokidx = [&](int i) -> int {
    if (LOCAL) return (wy * WSZ + (i >> 3)) * WW + wx * WSZ + (i & 7);
    return i;
  };

  size_t base = (size_t)b * LTK;
  const unsigned short* qp = qkv;
  const unsigned short* kp = qkv + (size_t)NTOK * CCH;
  const unsigned short* vp = qkv + (size_t)2 * NTOK * CCH;
  int hb = h * DH;

  // Q fragments: A-layout, two 32-wide d slices
  AF qf[2];
  {
    int qrow = tokidx(q0 + l15);
    size_t qoff = (base + qrow) * (size_t)CCH + hb;
#pragma unroll
    for (int kh = 0; kh < 2; ++kh)
#pragma unroll
      for (int v = 0; v < 8; ++v)
        qf[kh].u[v] = *(const unsigned*)&qp[qoff + kh * 32 + kpair(v, half)];
  }

  float mst[8], lst[8], alpha[8];
  v8f o[4] = {};
#pragma unroll
  for (int r = 0; r < 8; ++r) { mst[r] = -1e30f; lst[r] = 0.f; }

  for (int kc = 0; kc < nkeys; kc += 32) {
    if (!LOCAL) {
      __syncthreads();  // previous chunk fully consumed
      int key = tid >> 3, dg = (tid & 7) * 8;
      size_t roff = (base + kc + key) * (size_t)CCH + hb + dg;
      cp16_to_lds(&Ks[key][dg], &kp[roff]);          // async K stage
      uint4 vv = *(const uint4*)&vp[roff];           // V stage, transposed
      const unsigned short* vs = (const unsigned short*)&vv;
#pragma unroll
      for (int j = 0; j < 8; ++j) Vt[dg + j][key] = vs[j];
      async_fence();
      __syncthreads();
    }

    // ---- S = Q K^T for 32 keys (two 16-key n-tiles) ----
    v8f s0 = {}, s1 = {};
#pragma unroll
    for (int nt = 0; nt < 2; ++nt) {
#pragma unroll
      for (int kh = 0; kh < 2; ++kh) {
        AF bf;
        if (LOCAL) {
          int ktok = tokidx(kc + nt * 16 + l15);
          size_t krow = (base + ktok) * (size_t)CCH + hb;
#pragma unroll
          for (int v = 0; v < 8; ++v)
            bf.u[v] = *(const unsigned*)&kp[krow + kh * 32 + kpair(v, half)];
        } else {
#pragma unroll
          for (int v = 0; v < 8; ++v)
            bf.u[v] = *(const unsigned*)&Ks[nt * 16 + l15][kh * 32 + kpair(v, half)];
        }
        if (nt == 0) s0 = wmma_bf16(qf[kh], bf, s0);
        else         s1 = wmma_bf16(qf[kh], bf, s1);
      }
    }
    // fg key mask (global only); per-column, shared by all rows of this lane
    float km0 = 0.f, km1 = 0.f;
    if (!LOCAL) {
      km0 = (sel[base + kc + l15] > 1.5f) ? 0.f : -1e9f;
      km1 = (sel[base + kc + 16 + l15] > 1.5f) ? 0.f : -1e9f;
    }
    // ---- online softmax per row (row lives across 16 lanes of a half) ----
#pragma unroll
    for (int r = 0; r < 8; ++r) {
      float a0 = s0[r] * 0.125f + km0;
      float a1 = s1[r] * 0.125f + km1;
      float cm = fmaxf(a0, a1);
      cm = fmaxf(cm, __shfl_xor(cm, 1)); cm = fmaxf(cm, __shfl_xor(cm, 2));
      cm = fmaxf(cm, __shfl_xor(cm, 4)); cm = fmaxf(cm, __shfl_xor(cm, 8));
      float nm = fmaxf(mst[r], cm);
      float al = __expf(mst[r] - nm);
      float p0 = __expf(a0 - nm), p1 = __expf(a1 - nm);
      float rs = p0 + p1;
      rs += __shfl_xor(rs, 1); rs += __shfl_xor(rs, 2);
      rs += __shfl_xor(rs, 4); rs += __shfl_xor(rs, 8);
      lst[r] = lst[r] * al + rs;
      mst[r] = nm; alpha[r] = al;
      int m = r + 8 * half;
      pbuf[wid][m][l15]      = f2bf(p0);
      pbuf[wid][m][16 + l15] = f2bf(p1);
    }
#pragma unroll
    for (int nt = 0; nt < 4; ++nt)
#pragma unroll
      for (int r = 0; r < 8; ++r) o[nt][r] *= alpha[r];

    asm volatile("s_wait_dscnt 0" ::: "memory");  // wave-local LDS relayout fence

    // ---- P (A-frag) from LDS, then O += P * V ----
    AF pf;
#pragma unroll
    for (int v = 0; v < 8; ++v)
      pf.u[v] = *(const unsigned*)&pbuf[wid][l15][kpair(v, half)];
#pragma unroll
    for (int nt = 0; nt < 4; ++nt) {
      AF vf;
      if (LOCAL) {
        int dcol = hb + nt * 16 + l15;
#pragma unroll
        for (int v = 0; v < 8; ++v) {
          int k0 = kpair(v, half);
          unsigned lo = vp[(base + tokidx(kc + k0)) * (size_t)CCH + dcol];
          unsigned hi = vp[(base + tokidx(kc + k0 + 1)) * (size_t)CCH + dcol];
          vf.u[v] = lo | (hi << 16);
        }
      } else {
#pragma unroll
        for (int v = 0; v < 8; ++v)
          vf.u[v] = *(const unsigned*)&Vt[nt * 16 + l15][kpair(v, half)];
      }
      o[nt] = wmma_bf16(pf, vf, o[nt]);
    }
  }

  // ---- normalize and scatter ctx (bf16) ----
#pragma unroll
  for (int r = 0; r < 8; ++r) {
    int m = r + 8 * half;
    int qtok = tokidx(q0 + m);
    float inv = 1.f / lst[r];
#pragma unroll
    for (int nt = 0; nt < 4; ++nt)
      ctx[(base + qtok) * (size_t)CCH + hb + nt * 16 + l15] = f2bf(o[nt][r] * inv);
  }
}

// ---------------- scatter-back select + residual ----------------
__global__ void __launch_bounds__(256) combine_kernel(
    const float* __restrict__ x, const float* __restrict__ gout,
    const float* __restrict__ lout, const float* __restrict__ sel,
    float* __restrict__ x1, int total) {
  int i = blockIdx.x * 256 + threadIdx.x;
  if (i >= total) return;
  float sv = sel[i / CCH];
  float xn = (sv > 1.5f) ? gout[i] : ((sv > 0.5f) ? lout[i] : 0.f);
  x1[i] = x[i] + xn;
}

// ---------------- host launch ----------------
extern "C" void kernel_launch(void* const* d_in, const int* in_sizes, int n_in,
                              void* d_out, int out_size, void* d_ws, size_t ws_size,
                              hipStream_t stream) {
  (void)in_sizes; (void)n_in; (void)out_size; (void)ws_size;
  const float* x      = (const float*)d_in[0];
  const float* maskp  = (const float*)d_in[1];
  const float* wqkvg  = (const float*)d_in[2];
  const float* bqkvg  = (const float*)d_in[3];
  const float* wog    = (const float*)d_in[4];
  const float* bog    = (const float*)d_in[5];
  const float* wqkvl  = (const float*)d_in[6];
  const float* bqkvl  = (const float*)d_in[7];
  const float* wol    = (const float*)d_in[8];
  const float* bol    = (const float*)d_in[9];
  const float* ln1w   = (const float*)d_in[10];
  const float* ln1b   = (const float*)d_in[11];
  const float* ln2w   = (const float*)d_in[12];
  const float* ln2b   = (const float*)d_in[13];
  const float* wfc1   = (const float*)d_in[14];
  const float* bfc1   = (const float*)d_in[15];
  const float* wfc2   = (const float*)d_in[16];
  const float* bfc2   = (const float*)d_in[17];

  const size_t CC = (size_t)CCH * CCH;
  const size_t C3 = 3 * CC;
  const size_t CM = (size_t)CCH * MFF;
  const size_t NC = (size_t)NTOK * CCH;
  const size_t NM = (size_t)NTOK * MFF;

  size_t off = 0;
  auto take = [&](size_t bytes) -> char* {
    char* p = (char*)d_ws + off;
    off = (off + bytes + 255) & ~(size_t)255;
    return p;
  };
  unsigned short* wqkvg16 = (unsigned short*)take(C3 * 2);
  unsigned short* wog16   = (unsigned short*)take(CC * 2);
  unsigned short* wqkvl16 = (unsigned short*)take(C3 * 2);
  unsigned short* wol16   = (unsigned short*)take(CC * 2);
  unsigned short* wfc1_16 = (unsigned short*)take(CM * 2);
  unsigned short* wfc2_16 = (unsigned short*)take(CM * 2);
  float*          sel     = (float*)take((size_t)NTOK * 4);
  unsigned short* xn16    = (unsigned short*)take(NC * 2);
  unsigned short* qkvg16  = (unsigned short*)take(3 * NC * 2);
  unsigned short* qkvl16  = (unsigned short*)take(3 * NC * 2);
  unsigned short* ctxg16  = (unsigned short*)take(NC * 2);
  unsigned short* ctxl16  = (unsigned short*)take(NC * 2);
  float*          gout    = (float*)take(NC * 4);
  float*          lout    = (float*)take(NC * 4);
  float*          x1      = (float*)take(NC * 4);
  unsigned short* h16     = (unsigned short*)take(NC * 2);
  unsigned short* a16     = (unsigned short*)take(NM * 2);

  auto cvt = [&](const float* src, unsigned short* dst, size_t n) {
    cvt_bf16_kernel<<<(unsigned)((n + 255) / 256), 256, 0, stream>>>(src, dst, (int)n);
  };
  cvt(wqkvg, wqkvg16, C3);
  cvt(wog,   wog16,   CC);
  cvt(wqkvl, wqkvl16, C3);
  cvt(wol,   wol16,   CC);
  cvt(wfc1,  wfc1_16, CM);
  cvt(wfc2,  wfc2_16, CM);

  sel_kernel<<<BB * 36, 64, 0, stream>>>(maskp, sel);
  ln_kernel<<<NTOK, 256, 0, stream>>>(x, ln1w, ln1b, xn16);

  // QKV projections (3 each for global / local weights)
  for (int t = 0; t < 3; ++t) {
    gemm_bf16_wmma<<<dim3(CCH / BN, NTOK / BM), 256, 0, stream>>>(
        xn16, wqkvg16 + t * CC, bqkvg + t * CCH, nullptr,
        nullptr, qkvg16 + t * NC, NTOK, CCH, CCH, 0);
    gemm_bf16_wmma<<<dim3(CCH / BN, NTOK / BM), 256, 0, stream>>>(
        xn16, wqkvl16 + t * CC, bqkvl + t * CCH, nullptr,
        nullptr, qkvl16 + t * NC, NTOK, CCH, CCH, 0);
  }

  // attention: global (fg key mask) and windowed local
  attn_kernel<false><<<dim3(LTK / 16 / 8, NHD, BB), 256, 0, stream>>>(qkvg16, sel, ctxg16);
  attn_kernel<true><<<(BB * 36 * NHD * 4) / 8, 256, 0, stream>>>(qkvl16, sel, ctxl16);

  // output projections
  gemm_bf16_wmma<<<dim3(CCH / BN, NTOK / BM), 256, 0, stream>>>(
      ctxg16, wog16, bog, nullptr, gout, nullptr, NTOK, CCH, CCH, 0);
  gemm_bf16_wmma<<<dim3(CCH / BN, NTOK / BM), 256, 0, stream>>>(
      ctxl16, wol16, bol, nullptr, lout, nullptr, NTOK, CCH, CCH, 0);

  // scatter-back + residual -> x1
  combine_kernel<<<(unsigned)((NC + 255) / 256), 256, 0, stream>>>(
      x, gout, lout, sel, x1, (int)NC);

  // LN2 -> MLP (fc1 + exact GELU fused; fc2 + bias + residual -> d_out)
  ln_kernel<<<NTOK, 256, 0, stream>>>(x1, ln2w, ln2b, h16);
  gemm_bf16_wmma<<<dim3(MFF / BN, NTOK / BM), 256, 0, stream>>>(
      h16, wfc1_16, bfc1, nullptr, nullptr, a16, NTOK, MFF, CCH, 1);
  gemm_bf16_wmma<<<dim3(CCH / BN, NTOK / BM), 256, 0, stream>>>(
      a16, wfc2_16, bfc2, x1, (float*)d_out, nullptr, NTOK, CCH, MFF, 0);
}
